// AttentionModule_11012296147616
// MI455X (gfx1250) — matrix operand order repaired and checked by reading, not compiled, not run on hardware
//
#include <hip/hip_runtime.h>
#include <math.h>

// ---------------------------------------------------------------------------
// Problem constants: B=512, L=32, M=48, SH=1024, TH=4096, D=512
// ---------------------------------------------------------------------------
#define BATCH 512
#define LQ 32
#define MK 48
#define DDIM 512
#define NT (DDIM / 16)      // 32 n-tiles of 16
#define BLOCK 256           // attention kernel block (8 waves)
#define PBLOCK 512          // projection kernel block (16 waves)
#define ROWS_PER_WG 64      // GEMM row tile (4 WMMA row-tiles)

typedef __attribute__((ext_vector_type(16))) __bf16        v16bf;
typedef __attribute__((ext_vector_type(8)))  __bf16        v8bf;
typedef __attribute__((ext_vector_type(4)))  __bf16        v4bf;
typedef __attribute__((ext_vector_type(8)))  float         v8f;
typedef __attribute__((ext_vector_type(4)))  float         v4f;
typedef __attribute__((ext_vector_type(8)))  unsigned int  v8u;
typedef __attribute__((ext_vector_type(4)))  unsigned int  v4u;

__device__ __forceinline__ float gelu_exact(float x) {
    return 0.5f * x * (1.0f + erff(x * 0.70710678118654752f));
}

// ---------------------------------------------------------------------------
// WMMA fragment helpers (wave32; A and B both hold k = 16*(lane/16) + s in element s).
// ---------------------------------------------------------------------------
// A 16x32 bf16 from LDS row-major [rows][32].
__device__ __forceinline__ v16bf load_a_frag_lds(const __bf16* a, int rowBase, int lane) {
    int m = rowBase + (lane & 15);
    int kbase = (lane >> 4) * 16;
    const __bf16* p = a + m * 32 + kbase;
    v4u lo = *(const v4u*)(p);        // K kb..kb+7
    v4u hi = *(const v4u*)(p + 8);    // K kb+8..kb+15
    v8u raw;
    raw[0] = lo[0]; raw[1] = lo[1]; raw[2] = lo[2]; raw[3] = lo[3];
    raw[4] = hi[0]; raw[5] = hi[1]; raw[6] = hi[2]; raw[7] = hi[3];
    return __builtin_bit_cast(v16bf, raw);
}

// B tile as k^T from bf16 LDS [N rows][DDIM].
__device__ __forceinline__ v16bf load_bT_frag_lds(const __bf16* ks, int k0, int n0, int lane) {
    int n = n0 + (lane & 15);
    int kb = k0 + ((lane >> 4) << 4);
    const v8u* p = (const v8u*)(ks + (size_t)n * DDIM + kb);
    return __builtin_bit_cast(v16bf, *p);
}

// ---------------------------------------------------------------------------
// Kernel 0: pack W (f32 [Kdim x 512]) into bf16 fragment-native layout:
// Wp[((kt*NT + nt)*32 + lane)*16 + e] — a B fragment is 32 contiguous bytes/lane.
// ---------------------------------------------------------------------------
__global__ void pack_w_bf16(const float* __restrict__ W, __bf16* __restrict__ Wp, int Kdim) {
    int idx  = blockIdx.x * blockDim.x + threadIdx.x;
    int lane = idx & 31;
    int tile = idx >> 5;
    int tiles_total = (Kdim >> 5) * NT;
    if (tile >= tiles_total) return;
    int nt = tile & (NT - 1);
    int kt = tile / NT;
    int n  = nt * 16 + (lane & 15);
    int kb = kt * 32 + ((lane >> 4) << 4);
    v16bf f;
#pragma unroll
    for (int e = 0; e < 16; ++e)
        f[e] = (__bf16)W[(size_t)(kb + e) * DDIM + n];
    *(v8u*)(Wp + ((size_t)tile * 32 + lane) * 16) = __builtin_bit_cast(v8u, f);
}

// ---------------------------------------------------------------------------
// Kernel 1: out = LayerNorm(GELU(concat(xo,xd) @ W + bias)), packed bf16 W.
// 64 rows x 512 cols per WG; 16 waves, wave w owns cols [32w, 32w+32).
// Double-buffered A tile; one barrier per K step.
// ---------------------------------------------------------------------------
__global__ void __launch_bounds__(PBLOCK, 1)
proj_gelu_ln_kernel(const float* __restrict__ xo,
                    const float* __restrict__ xd,
                    const __bf16* __restrict__ Wp,
                    const float* __restrict__ bias,
                    const float* __restrict__ gamma,
                    const float* __restrict__ beta,
                    float* __restrict__ out,
                    int Kdim, int half) {
    extern __shared__ char smem[];
    __bf16* aT0  = (__bf16*)smem;                           // 64 x 32 bf16 (4 KB)
    __bf16* aT1  = (__bf16*)(smem + 4096);                  // second buffer (4 KB)
    float*  pSum = (float*)(smem + 8192);                   // [64][16] (4 KB)
    float*  pSq  = (float*)(smem + 8192 + 4096);            // [64][16] (4 KB)
    float*  muA  = (float*)(smem + 8192 + 8192);            // [64]
    float*  invA = (float*)(smem + 8192 + 8192 + 256);      // [64]

    const int t    = threadIdx.x;
    const int lane = t & 31;
    const int w    = t >> 5;               // wave 0..15
    const int row0 = blockIdx.x * ROWS_PER_WG;
    const int col0 = w * 32;

    // Staging: thread t stages 4 consecutive features of one row (one b128 load).
    const int stR = t >> 3;                // row 0..63
    const int stK = (t & 7) * 4;           // k offset 0..28

    const size_t stRow = (size_t)(row0 + stR);

    v8f acc[4][2] = {};

    // Packed-B walking pointer: tile (kt, nt0 = 2w); +NT*32*16 elements per K step.
    const __bf16* bPtr = Wp + ((size_t)(w * 2) * 32 + lane) * 16;
    const size_t  bStride = (size_t)NT * 32 * 16;   // elements per K step

    const int ksteps = Kdim >> 5;

    // prologue: stage K-step 0 into aT0
    {
        int f = stK;
        const float* src = (f < half) ? (xo + stRow * half + f)
                                      : (xd + stRow * half + (f - half));
        v4f x = *(const v4f*)(src);
        v4bf o;
#pragma unroll
        for (int e = 0; e < 4; ++e) o[e] = (__bf16)x[e];
        *(v4bf*)(aT0 + stR * 32 + stK) = o;
    }
    __syncthreads();

    for (int kt = 0; kt < ksteps; ++kt) {
        const __bf16* aCur = (kt & 1) ? aT1 : aT0;
        __bf16*       aNxt = (kt & 1) ? aT0 : aT1;

        // stage next K step (writes the other buffer; no barrier needed yet)
        if (kt + 1 < ksteps) {
            int f = ((kt + 1) << 5) + stK;
            const float* src = (f < half) ? (xo + stRow * half + f)
                                          : (xd + stRow * half + (f - half));
            v4f x = *(const v4f*)(src);
            v4bf o;
#pragma unroll
            for (int e = 0; e < 4; ++e) o[e] = (__bf16)x[e];
            *(v4bf*)(aNxt + stR * 32 + stK) = o;
            // prefetch next K-step's packed B block (global_prefetch_b8)
            __builtin_prefetch((const void*)(bPtr + bStride), 0, 1);
        }

        // B fragments for this wave's two column tiles (2 x 32B contiguous loads)
        v16bf bF0 = __builtin_bit_cast(v16bf, *(const v8u*)(bPtr));
        v16bf bF1 = __builtin_bit_cast(v16bf, *(const v8u*)(bPtr + 512));
        bPtr += bStride;

#pragma unroll
        for (int rt = 0; rt < 4; ++rt) {
            v16bf aF = load_a_frag_lds(aCur, rt * 16, lane);
            acc[rt][0] = __builtin_amdgcn_wmma_f32_16x16x32_bf16(
                false, aF, false, bF0, (short)0, acc[rt][0], false, false);
            acc[rt][1] = __builtin_amdgcn_wmma_f32_16x16x32_bf16(
                false, aF, false, bF1, (short)0, acc[rt][1], false, false);
        }
        __syncthreads();
    }

    // ---- epilogue: bias + GELU in registers ----
    const int mhalf = (lane >> 4) << 3;
    float gam[2], bet[2], bn[2];
#pragma unroll
    for (int ct = 0; ct < 2; ++ct) {
        int n = col0 + ct * 16 + (lane & 15);
        bn[ct]  = bias[n];
        gam[ct] = gamma[n];
        bet[ct] = beta[n];
    }
#pragma unroll
    for (int rt = 0; rt < 4; ++rt)
#pragma unroll
        for (int ct = 0; ct < 2; ++ct)
#pragma unroll
            for (int j = 0; j < 8; ++j)
                acc[rt][ct][j] = gelu_exact(acc[rt][ct][j] + bn[ct]);

    // ---- per-row partial sums over this wave's 32 cols ----
#pragma unroll
    for (int rt = 0; rt < 4; ++rt) {
#pragma unroll
        for (int j = 0; j < 8; ++j) {
            float s = 0.f, s2 = 0.f;
#pragma unroll
            for (int ct = 0; ct < 2; ++ct) {
                float v = acc[rt][ct][j];
                s += v; s2 += v * v;
            }
#pragma unroll
            for (int off = 1; off <= 8; off <<= 1) {
                s  += __shfl_xor(s,  off, 32);
                s2 += __shfl_xor(s2, off, 32);
            }
            if ((lane & 15) == 0) {
                int m = rt * 16 + mhalf + j;
                pSum[m * 16 + w] = s;
                pSq[m * 16 + w]  = s2;
            }
        }
    }
    __syncthreads();

    if (t < ROWS_PER_WG) {
        float s = 0.f, s2 = 0.f;
#pragma unroll
        for (int i = 0; i < 16; ++i) { s += pSum[t * 16 + i]; s2 += pSq[t * 16 + i]; }
        float mu  = s * (1.0f / DDIM);
        float var = s2 * (1.0f / DDIM) - mu * mu;
        muA[t]  = mu;
        invA[t] = rsqrtf(var + 1e-5f);
    }
    __syncthreads();

    // ---- normalize + store ----
#pragma unroll
    for (int rt = 0; rt < 4; ++rt) {
#pragma unroll
        for (int j = 0; j < 8; ++j) {
            int m = rt * 16 + mhalf + j;
            float mu  = muA[m];
            float inv = invA[m];
            size_t rowg = (size_t)(row0 + m) * DDIM;
#pragma unroll
            for (int ct = 0; ct < 2; ++ct) {
                int n = col0 + ct * 16 + (lane & 15);
                out[rowg + n] = (acc[rt][ct][j] - mu) * inv * gam[ct] + bet[ct];
            }
        }
    }
}

// ---------------------------------------------------------------------------
// Kernel 2 (per batch b): scores = q@k^T*scale + depth_bias, softmax, z=alpha@basis
// ---------------------------------------------------------------------------
__global__ void __launch_bounds__(BLOCK)
attn_kernel(const float* __restrict__ q,
            const float* __restrict__ k,
            const float* __restrict__ basis,
            float* __restrict__ z,
            float* __restrict__ alpha) {
    extern __shared__ char smem[];
    __bf16* qs = (__bf16*)smem;                  // 32 x 512 bf16 (32 KB)
    __bf16* ks = qs + LQ * DDIM;                 // 48 x 512 bf16 (48 KB)
    float*  sc = (float*)(smem + (size_t)(LQ + MK) * DDIM * 2); // 32 x 48 f32

    const int b    = blockIdx.x;
    const int t    = threadIdx.x;
    const int lane = t & 31;
    const int w    = t >> 5;

    const float* qb = q + (size_t)b * LQ * DDIM;
    const float* kb = k + (size_t)b * MK * DDIM;

    // stage q,k as bf16 into LDS: 8 consecutive f32 per thread-chunk.
#pragma unroll
    for (int i = 0; i < (LQ * DDIM) / (BLOCK * 8); ++i) {
        int off = (t + i * BLOCK) * 8;
        v4f x0 = *(const v4f*)(qb + off);
        v4f x1 = *(const v4f*)(qb + off + 4);
        v8bf o;
#pragma unroll
        for (int e = 0; e < 4; ++e) { o[e] = (__bf16)x0[e]; o[4 + e] = (__bf16)x1[e]; }
        *(v8bf*)(qs + off) = o;
    }
#pragma unroll
    for (int i = 0; i < (MK * DDIM) / (BLOCK * 8); ++i) {
        int off = (t + i * BLOCK) * 8;
        v4f x0 = *(const v4f*)(kb + off);
        v4f x1 = *(const v4f*)(kb + off + 4);
        v8bf o;
#pragma unroll
        for (int e = 0; e < 4; ++e) { o[e] = (__bf16)x0[e]; o[4 + e] = (__bf16)x1[e]; }
        *(v8bf*)(ks + off) = o;
    }
    __syncthreads();

    // 6 score tiles (2 row-tiles x 3 col-tiles), one per wave.
    if (w < 6) {
        const int rt = w / 3, ct = w % 3;
        v8f acc = {};
        const int m   = rt * 16 + (lane & 15);
        const int kbs = (lane >> 4) * 16;
        for (int k0 = 0; k0 < DDIM; k0 += 32) {
            const __bf16* p = qs + m * DDIM + k0 + kbs;
            v4u lo = *(const v4u*)(p);
            v4u hi = *(const v4u*)(p + 8);
            v8u raw;
            raw[0] = lo[0]; raw[1] = lo[1]; raw[2] = lo[2]; raw[3] = lo[3];
            raw[4] = hi[0]; raw[5] = hi[1]; raw[6] = hi[2]; raw[7] = hi[3];
            v16bf aF = __builtin_bit_cast(v16bf, raw);
            v16bf bF = load_bT_frag_lds(ks, k0, ct * 16, lane);
            acc = __builtin_amdgcn_wmma_f32_16x16x32_bf16(
                false, aF, false, bF, (short)0, acc, false, false);
        }
        const float scale = 0.044194173824159216f;   // 1/sqrt(512)
        const int mh = (lane >> 4) << 3;
        const int n  = ct * 16 + (lane & 15);
        const float nfrac = (float)n * (1.0f / (MK - 1));
#pragma unroll
        for (int j = 0; j < 8; ++j) {
            int mm = rt * 16 + mh + j;
            float lfrac = (float)mm * (1.0f / (LQ - 1));
            sc[mm * MK + n] = acc[j] * scale - fabsf(lfrac - nfrac);
        }
    }
    __syncthreads();

    // softmax over M=48, one thread per row.
    if (t < LQ) {
        float mx = -1e30f;
        for (int m2 = 0; m2 < MK; ++m2) mx = fmaxf(mx, sc[t * MK + m2]);
        float sum = 0.f;
        for (int m2 = 0; m2 < MK; ++m2) sum += __expf(sc[t * MK + m2] - mx);
        float inv = 1.0f / sum;
        size_t ao = (size_t)b * LQ * MK + (size_t)t * MK;
        for (int m2 = 0; m2 < MK; ++m2) {
            float a2 = __expf(sc[t * MK + m2] - mx) * inv;
            sc[t * MK + m2] = a2;
            alpha[ao + m2] = a2;
        }
    }
    __syncthreads();

    // z = alpha @ basis (f32 VALU, trivial FLOPs)
    for (int e = t; e < LQ * DDIM; e += BLOCK) {
        int l = e >> 9;
        int dcol = e & (DDIM - 1);
        float sum = 0.f;
#pragma unroll 4
        for (int m2 = 0; m2 < MK; ++m2)
            sum = fmaf(sc[l * MK + m2], basis[m2 * DDIM + dcol], sum);
        z[(size_t)b * LQ * DDIM + e] = sum;
    }
}

// ---------------------------------------------------------------------------
extern "C" void kernel_launch(void* const* d_in, const int* in_sizes, int n_in,
                              void* d_out, int out_size, void* d_ws, size_t ws_size,
                              hipStream_t stream) {
    const float* student_o = (const float*)d_in[0];
    const float* student_d = (const float*)d_in[1];
    const float* teacher_o = (const float*)d_in[2];
    const float* teacher_d = (const float*)d_in[3];
    const float* Wq    = (const float*)d_in[4];
    const float* bq    = (const float*)d_in[5];
    const float* gq    = (const float*)d_in[6];
    const float* betaq = (const float*)d_in[7];
    const float* Wk    = (const float*)d_in[8];
    const float* bk    = (const float*)d_in[9];
    const float* gk    = (const float*)d_in[10];
    const float* betak = (const float*)d_in[11];
    const float* basis = (const float*)d_in[12];

    // workspace carve-up (bytes)
    char* ws = (char*)d_ws;
    __bf16* WqP = (__bf16*)ws;                                   // 2 MB
    __bf16* WkP = (__bf16*)(ws + (size_t)2048 * 512 * 2);        // 8 MB
    float*  qbuf = (float*)(ws + (size_t)2048 * 512 * 2 + (size_t)8192 * 512 * 2);
    float*  kbuf = qbuf + (size_t)BATCH * LQ * DDIM;

    float* z     = (float*)d_out;                                // [B, L, D]
    float* alpha = z + (size_t)BATCH * LQ * DDIM;                // [B, L, M]

    // ---- pack weights to bf16 fragment layout ----
    {
        int tilesQ = (2048 / 32) * NT;
        int tilesK = (8192 / 32) * NT;
        hipLaunchKernelGGL(pack_w_bf16, dim3((tilesQ * 32) / BLOCK), dim3(BLOCK), 0, stream,
                           Wq, WqP, 2048);
        hipLaunchKernelGGL(pack_w_bf16, dim3((tilesK * 32) / BLOCK), dim3(BLOCK), 0, stream,
                           Wk, WkP, 8192);
    }

    // ---- fused proj+GELU+LN ----
    size_t lds1 = 4096 * 2 + 4096 * 2 + 256 + 256;  // 2x aTile + partials + mu/inv
    hipLaunchKernelGGL(proj_gelu_ln_kernel,
                       dim3((BATCH * LQ) / ROWS_PER_WG), dim3(PBLOCK), lds1, stream,
                       student_o, student_d, WqP, bq, gq, betaq, qbuf, 2048, 1024);
    hipLaunchKernelGGL(proj_gelu_ln_kernel,
                       dim3((BATCH * MK) / ROWS_PER_WG), dim3(PBLOCK), lds1, stream,
                       teacher_o, teacher_d, WkP, bk, gk, betak, kbuf, 8192, 4096);

    // ---- attention ----
    size_t lds2 = (size_t)(LQ + MK) * DDIM * 2 + (size_t)LQ * MK * 4;
    hipLaunchKernelGGL(attn_kernel, dim3(BATCH), dim3(BLOCK), lds2, stream,
                       qbuf, kbuf, basis, z, alpha);
}
